// InceptionResNet_36043365548505
// MI455X (gfx1250) — matrix operand order repaired and checked by reading, not compiled, not run on hardware
//
#include <hip/hip_runtime.h>

// CDNA5 / gfx1250 sparse-conv Inception-ResNet block.
// bf16 features + v_wmma_f32_16x16x32_bf16; gathered rows load directly into
// the WMMA A-fragment layout (2x global_load_b128 per lane), weights
// pre-transposed to [k][cout][cin] so B is also 2x b128 per lane.
// Neighbor masking via address-select to a zero row (2 cndmask/tap instead of
// 8 cndmask/fragment) to keep VALU pressure out of the WMMA loop.

typedef __attribute__((ext_vector_type(16))) __bf16        v16bf;
typedef __attribute__((ext_vector_type(8)))  float         v8f;
typedef __attribute__((ext_vector_type(4)))  unsigned int  v4u;

typedef unsigned short u16;
typedef unsigned int   u32;

__device__ __forceinline__ u16 f2bf(float f) {
    u32 u = __builtin_bit_cast(u32, f);
    u = (u + 0x7FFFu + ((u >> 16) & 1u)) >> 16;  // round-to-nearest-even
    return (u16)u;
}

// ---------------- fp32 -> bf16 streaming convert ----------------
__global__ __launch_bounds__(256) void cvt_bf16_kernel(const float* __restrict__ src,
                                                       u16* __restrict__ dst, long n) {
    long i = (long)blockIdx.x * blockDim.x + threadIdx.x;
    const long stride = (long)gridDim.x * blockDim.x;
    for (; i < n; i += stride) dst[i] = f2bf(src[i]);
}

// ---------------- zero a small scratch region (the masked-gather zero row) ----------------
__global__ __launch_bounds__(256) void zfill_kernel(u16* __restrict__ dst, int n) {
    int i = blockIdx.x * blockDim.x + threadIdx.x;
    if (i < n) dst[i] = 0;
}

// ------------- weight transpose [k][cin][cout] -> bf16 [k][cout][cin] -------------
__global__ __launch_bounds__(256) void wprep_kernel(const float* __restrict__ W,
                                                    u16* __restrict__ Wt,
                                                    int NK, int Cin, int Cout) {
    const int total = NK * Cin * Cout;
    for (int i = blockIdx.x * blockDim.x + threadIdx.x; i < total;
         i += gridDim.x * blockDim.x) {
        int k  = i / (Cin * Cout);
        int r  = i - k * (Cin * Cout);
        int ci = r / Cout;
        int co = r - ci * Cout;
        Wt[((long)k * Cout + co) * Cin + ci] = f2bf(W[i]);
    }
}

// ------------- gathered GEMM: out[i] = sum_k mask[i,k] * F[nbr[i,k]] @ W[k] -------------
// One wave per 16-point tile.  A fragment = gathered feature rows (row-major
// [point][cin] matches WMMA 16-bit A striping exactly).  B fragment = bf16
// transposed weights (column n = lane%16, consecutive K).  EXEC is all-ones
// at every WMMA; masked taps gather from a zero row instead of zeroing regs.
template<int CIN, int COUT, int NK, bool RELU, bool FINAL>
__global__ __launch_bounds__(256) void gemm_gather_kernel(
    const u16* __restrict__ F,             // [npts][CIN] bf16 features
    const int* __restrict__ nbr,           // [npts][NK] (unused if NK==1)
    const unsigned char* __restrict__ mask,// [npts][NK] bool (unused if NK==1)
    const u16* __restrict__ zrow,          // [CIN] bf16 zeros (masked-gather target)
    const u16* __restrict__ Wt,            // [NK][COUT][CIN] bf16
    const float* __restrict__ bias,        // [COUT]
    const float* __restrict__ xres,        // [npts][128] residual (FINAL only)
    u16* __restrict__ outH,                // [npts][COUT] bf16 (if !FINAL)
    float* __restrict__ outF,              // [npts][128] fp32 (if FINAL)
    int colOfs, int npts)
{
    constexpr int NT = COUT / 16;  // N-tiles of 16 output channels
    constexpr int KS = CIN  / 32;  // K-slices of 32 input channels

    const int lane  = threadIdx.x & 31;
    const int wave  = blockIdx.x * (blockDim.x >> 5) + (threadIdx.x >> 5);
    const int tile0 = wave * 16;
    if (tile0 >= npts) return;     // uniform per wave

    const int  row = lane & 15;    // point row this lane feeds (A) / out channel (B,C)
    const bool hi  = lane >= 16;
    const int  p   = tile0 + row;
    const bool pv  = p < npts;
    const int  pc  = pv ? p : 0;

    v8f acc[NT] = {};

    union AB { v4u u[2]; v16bf v; };

    for (int k = 0; k < NK; ++k) {
        const u16* frow;
        if constexpr (NK == 1) {
            frow = pv ? (F + (long)pc * CIN) : zrow;
        } else {
            const int  idx = nbr[(long)pc * NK + k];
            const bool m   = pv && (mask[(long)pc * NK + k] != 0);
            // address-select: masked/invalid rows gather the L0-resident zero row
            frow = m ? (F + (long)idx * CIN) : zrow;
        }
        #pragma unroll
        for (int ks = 0; ks < KS; ++ks) {
            // A fragment: lane<16 -> K {0..7,16..23}, lane>=16 -> K {8..15,24..31}
            const int aoff = ks * 32 + (hi ? 8 : 0);
            AB a;
            a.u[0] = *(const v4u*)(frow + aoff);
            a.u[1] = *(const v4u*)(frow + aoff + 16);
            #pragma unroll
            for (int t = 0; t < NT; ++t) {
                // B fragment: column n = t*16+row, lane<16 -> K {0..7,16..23}, lane>=16 -> K {8..15,24..31}
                const u16* wrow = Wt + ((long)(k * COUT + t * 16 + row)) * CIN
                                     + ks * 32 + (hi ? 8 : 0);
                AB b;
                b.u[0] = *(const v4u*)(wrow);
                b.u[1] = *(const v4u*)(wrow + 16);
                acc[t] = __builtin_amdgcn_wmma_f32_16x16x32_bf16(
                    false, a.v, false, b.v, (short)0, acc[t], false, false);
            }
        }
    }

    // Epilogue: C/D layout lane<16 -> (M=j, N=lane); lane>=16 -> (M=j+8, N=lane-16)
    #pragma unroll
    for (int t = 0; t < NT; ++t) {
        const int   ch = t * 16 + row;
        const float bv = bias[ch];
        #pragma unroll
        for (int j = 0; j < 8; ++j) {
            const int pm = tile0 + j + (hi ? 8 : 0);
            if (pm < npts) {
                float v = acc[t][j] + bv;
                if constexpr (RELU) v = fmaxf(v, 0.0f);
                if constexpr (FINAL) {
                    const long o = (long)pm * 128 + colOfs + ch;
                    // NT hints: keep streamed fp32 I/O from evicting gather-hot bf16 L2 lines
                    __builtin_nontemporal_store(v + __builtin_nontemporal_load(&xres[o]),
                                                &outF[o]);
                } else {
                    outH[(long)pm * COUT + ch] = f2bf(v);
                }
            }
        }
    }
}

extern "C" void kernel_launch(void* const* d_in, const int* in_sizes, int n_in,
                              void* d_out, int out_size, void* d_ws, size_t ws_size,
                              hipStream_t stream)
{
    (void)n_in; (void)out_size; (void)ws_size;

    const float*         x    = (const float*)d_in[0];
    const int*           nbr  = (const int*)d_in[1];
    const unsigned char* mask = (const unsigned char*)d_in[2];  // jax bool = 1 byte
    const float* W00 = (const float*)d_in[3];  const float* b00 = (const float*)d_in[4];
    const float* W01 = (const float*)d_in[5];  const float* b01 = (const float*)d_in[6];
    const float* W02 = (const float*)d_in[7];  const float* b02 = (const float*)d_in[8];
    const float* W10 = (const float*)d_in[9];  const float* b10 = (const float*)d_in[10];
    const float* W11 = (const float*)d_in[11]; const float* b11 = (const float*)d_in[12];

    const int C = 128;
    const int N = in_sizes[0] / C;

    // carve scratch
    size_t off = 0;
    auto carve = [&](size_t bytes) -> void* {
        void* p = (char*)d_ws + off;
        off += (bytes + 255) & ~(size_t)255;
        return p;
    };
    u16* xh   = (u16*)carve((size_t)N * 128 * 2);   // bf16 copy of x
    u16* h0   = (u16*)carve((size_t)N * 32  * 2);
    u16* h0c  = (u16*)carve((size_t)N * 32  * 2);
    u16* h1   = (u16*)carve((size_t)N * 32  * 2);
    u16* Wt00 = (u16*)carve((size_t)1  * 32 * 128 * 2);
    u16* Wt01 = (u16*)carve((size_t)27 * 32 * 32  * 2);
    u16* Wt02 = (u16*)carve((size_t)1  * 64 * 32  * 2);
    u16* Wt10 = (u16*)carve((size_t)27 * 32 * 128 * 2);
    u16* Wt11 = (u16*)carve((size_t)27 * 64 * 32  * 2);
    u16* zrow = (u16*)carve((size_t)128 * 2);       // zero row for masked gathers
    float* outF = (float*)d_out;

    // weight transpose + convert (tiny) and zero-row init
    wprep_kernel<<<32,  256, 0, stream>>>(W00, Wt00, 1,  128, 32);
    wprep_kernel<<<64,  256, 0, stream>>>(W01, Wt01, 27, 32,  32);
    wprep_kernel<<<16,  256, 0, stream>>>(W02, Wt02, 1,  32,  64);
    wprep_kernel<<<128, 256, 0, stream>>>(W10, Wt10, 27, 128, 32);
    wprep_kernel<<<64,  256, 0, stream>>>(W11, Wt11, 27, 32,  64);
    zfill_kernel<<<1, 256, 0, stream>>>(zrow, 128);

    // features -> bf16
    cvt_bf16_kernel<<<4096, 256, 0, stream>>>(x, xh, (long)N * C);

    const int tiles  = (N + 15) / 16;
    const int blocks = (tiles + 7) / 8;   // 8 waves / 256-thread block

    // branch 0: 1x1 (128->32) -> relu
    gemm_gather_kernel<128, 32, 1,  true,  false><<<blocks, 256, 0, stream>>>(
        xh, nullptr, nullptr, zrow, Wt00, b00, nullptr, h0, nullptr, 0, N);
    // branch 0: 3x3 (32->32) -> relu
    gemm_gather_kernel<32,  32, 27, true,  false><<<blocks, 256, 0, stream>>>(
        h0, nbr, mask, zrow, Wt01, b01, nullptr, h0c, nullptr, 0, N);
    // branch 1: 3x3 (128->32) -> relu   (the bandwidth-dominant layer)
    gemm_gather_kernel<128, 32, 27, true,  false><<<blocks, 256, 0, stream>>>(
        xh, nbr, mask, zrow, Wt10, b10, nullptr, h1, nullptr, 0, N);
    // branch 0 tail: 1x1 (32->64) + bias + residual -> out[:, 0:64]
    gemm_gather_kernel<32,  64, 1,  false, true ><<<blocks, 256, 0, stream>>>(
        h0c, nullptr, nullptr, zrow, Wt02, b02, x, nullptr, outF, 0, N);
    // branch 1 tail: 3x3 (32->64) + bias + residual -> out[:, 64:128]
    gemm_gather_kernel<32,  64, 27, false, true ><<<blocks, 256, 0, stream>>>(
        h1, nbr, mask, zrow, Wt11, b11, x, nullptr, outF, 64, N);
}
